// LSTMModel_85469849191033
// MI455X (gfx1250) — matrix operand order repaired and checked by reading, not compiled, not run on hardware
//
#include <hip/hip_runtime.h>
#include <hip/hip_bf16.h>

// ---------------- problem constants ----------------
constexpr int kB   = 512;   // batch
constexpr int kF   = 64;    // features
constexpr int kH   = 128;   // hidden
constexpr int kT   = 1024;  // seq len
constexpr int kNW  = 8;     // waves per block
constexpr int kNT  = 256;   // threads per block
constexpr int kRows = 16;   // batch rows per block

// weight fragments pre-swizzled into LDS (bf16, WMMA B-layout)
constexpr int kFragG = 32 * 6;          // 32 gate n-tiles x 6 k-blocks (K=192)
constexpr int kFragD = 4 * 4;           // 4 dense n-tiles x 4 k-blocks (K=128)
constexpr int kFrag  = kFragG + kFragD; // 208 fragments x 1KB = 208KB

typedef __attribute__((ext_vector_type(16))) __bf16 v16bf;
typedef __attribute__((ext_vector_type(8)))  float  v8f;

// gfx1250 has a native v_tanh_f32 transcendental.
#if __has_builtin(__builtin_amdgcn_tanhf)
#define FAST_TANH(x) __builtin_amdgcn_tanhf(x)
#else
#define FAST_TANH(x) tanhf(x)
#endif
__device__ __forceinline__ float sigf(float x) {
  // sigmoid(x) = 0.5*tanh(x/2) + 0.5  -> one TRANS op instead of exp+rcp
  return fmaf(0.5f, FAST_TANH(0.5f * x), 0.5f);
}

__device__ __forceinline__ v8f splat8(float v) {
  v8f r;
#pragma unroll
  for (int i = 0; i < 8; ++i) r[i] = v;
  return r;
}

// A-matrix 16x32 bf16 fragment (ISA 7.12.2): lanes 0-15 -> M=lane, K groups
// {0-7,16-23}; lanes 16-31 -> M=lane-16, K groups {8-15,24-31}.
// Elements 0..7 and 8..15 are each 8 contiguous bf16 -> two ds_load_b128.
__device__ __forceinline__ v16bf load_a_frag(const __bf16* buf, int ldk, int kbase, int lane) {
  v16bf a;
  const int m  = lane & 15;
  const int hi = (lane >> 4) & 1;
#pragma unroll
  for (int e = 0; e < 16; ++e) {
    const int k = kbase + ((e >> 3) * 16) + hi * 8 + (e & 7);
    a[e] = buf[m * ldk + k];
  }
  return a;
}

// B-matrix fragment: pre-swizzled in LDS so each lane reads 32 contiguous bytes.
__device__ __forceinline__ v16bf load_b_frag(const __bf16* wf, int frag, int lane) {
  return *(const v16bf*)(wf + frag * 512 + lane * 16);
}

__device__ __forceinline__ v8f wmma_bf16(v16bf a, v16bf b, v8f c) {
  return __builtin_amdgcn_wmma_f32_16x16x32_bf16(false, a, false, b, (short)0, c, false, false);
}

extern "C" __global__ void __launch_bounds__(kNT, 1)
lstm_autoreg_kernel(const float* __restrict__ xg,
                    const int*   __restrict__ avail_g,
                    const int*   __restrict__ pred_g,
                    const float* __restrict__ Wih,
                    const float* __restrict__ Whh,
                    const float* __restrict__ bih,
                    const float* __restrict__ bhh,
                    const float* __restrict__ Wden,
                    const float* __restrict__ bden,
                    const float* __restrict__ initch,
                    float*       __restrict__ outg) {
  extern __shared__ __align__(128) char smem_raw[];
  __bf16* wfrag   = (__bf16*)smem_raw;             // 208 * 512 bf16  (208 KB)
  __bf16* hbuf0   = wfrag + kFrag * 512;           // 16 x 128 bf16 (ping)
  __bf16* hbuf1   = hbuf0 + kRows * kH;            // 16 x 128 bf16 (pong)
  __bf16* xbuf    = hbuf1 + kRows * kH;            // 16 x 64 bf16 (x_in, built in-place)
  int*    avail_s = (int*)(xbuf + kRows * kF);     // 16
  int*    pred_s  = avail_s + kRows;               // 16
  int*    redux   = pred_s + kRows;                // 1 (max predict_len)

  const int tid  = threadIdx.x;
  const int lane = tid & 31;
  const int w    = tid >> 5;           // wave id 0..7
  const int b0   = blockIdx.x * kRows; // this block's batch slice

  // ---------------- init ----------------
  if (tid == 0) redux[0] = 1;
  if (tid < kRows) {
    avail_s[tid] = avail_g[b0 + tid];
    pred_s[tid]  = pred_g[b0 + tid];
  }
  __syncthreads();
  for (int i = tid; i < kB; i += kNT) atomicMax(&redux[0], pred_g[i]);

  // Pre-swizzle all weights into WMMA B-fragment layout (bf16), once.
  // B layout: element e at lane L -> n = tile*16 + (L&15), k = kblk*32 + (e>>3)*16 + ((L>=16)?8:0) + (e&7)
  for (int frag = w; frag < kFrag; frag += kNW) {
    const int nl   = lane & 15;
    const int kb16 = ((lane >> 4) & 1) * 16;
#pragma unroll
    for (int e = 0; e < 16; ++e) {
      const int kl = ((e >> 3) * 16) + (kb16 >> 1) + (e & 7);
      float v;
      if (frag < kFragG) {
        const int nt = frag / 6, kb = frag % 6;
        const int col = nt * 16 + nl;       // gate column 0..511
        const int k   = kb * 32 + kl;       // 0..191 (F then H)
        v = (k < kF) ? Wih[col * kF + k] : Whh[col * kH + (k - kF)];
      } else {
        const int d  = frag - kFragG;
        const int wv = d >> 2, kb = d & 3;
        const int fcol = wv * 16 + nl;      // output feature 0..63
        const int k    = kb * 32 + kl;      // 0..127
        v = Wden[fcol * kH + k];
      }
      wfrag[frag * 512 + lane * 16 + e] = (__bf16)v;
    }
  }

  // initial h (init_ch[:, H:2H] broadcast over batch rows)
  for (int i = tid; i < kRows * kH; i += kNT)
    hbuf0[i] = (__bf16)initch[kH + (i & (kH - 1))];

  // initial x_in = x[:, :, 0] (feedback impossible at t=0)
  for (int idx = tid; idx < kRows * kF; idx += kNT) {
    const int m = idx >> 6, f = idx & (kF - 1);
    xbuf[idx] = (__bf16)xg[((size_t)(b0 + m) * kF + f) * kT];
  }

  // per-wave constants: this wave owns hidden columns [16w, 16w+16)
  const int n    = lane & 15;
  const int hi   = (lane >> 4) & 1;
  const int hcol = w * 16 + n;
  const float bias_i = bih[0 * kH + hcol] + bhh[0 * kH + hcol];
  const float bias_f = bih[1 * kH + hcol] + bhh[1 * kH + hcol];
  const float bias_g = bih[2 * kH + hcol] + bhh[2 * kH + hcol];
  const float bias_o = bih[3 * kH + hcol] + bhh[3 * kH + hcol];
  const float dbias  = (w < 4) ? bden[w * 16 + n] : 0.0f;

  v8f creg = splat8(initch[hcol]);  // c state, D-layout, persistent in VGPRs

  __syncthreads();
  const int max_idx = (redux[0] < kT) ? redux[0] : kT;

  // Hoist ALL loop-invariant B-fragments into registers (compiler uses
  // s_set_vgpr_msb to address >256 VGPRs; occupancy is LDS-bound anyway):
  // 24 gate fragments (192 VGPRs) + 4 dense fragments (32 VGPRs).
  v16bf wgi[6], wgf[6], wgg[6], wgo[6];
#pragma unroll
  for (int kb = 0; kb < 6; ++kb) {
    wgi[kb] = load_b_frag(wfrag, (0 * 8 + w) * 6 + kb, lane);
    wgf[kb] = load_b_frag(wfrag, (1 * 8 + w) * 6 + kb, lane);
    wgg[kb] = load_b_frag(wfrag, (2 * 8 + w) * 6 + kb, lane);
    wgo[kb] = load_b_frag(wfrag, (3 * 8 + w) * 6 + kb, lane);
  }
  v16bf wd[4];
#pragma unroll
  for (int kb = 0; kb < 4; ++kb)
    wd[kb] = load_b_frag(wfrag, kFragG + (w & 3) * 4 + kb, lane);

  // ---------------- serial time loop (2 barriers / step) ----------------
  for (int t = 0; t < max_idx; ++t) {
    const __bf16* hcur = (t & 1) ? hbuf1 : hbuf0;
    __bf16*       hnxt = (t & 1) ? hbuf0 : hbuf1;

    // phase A: gates = [x_in | h] x W^T (K=192, 24 WMMAs, B all in VGPRs),
    // then pointwise LSTM cell update -> hnxt (bf16, ping-pong).
    v8f ai = splat8(bias_i), af = splat8(bias_f);
    v8f ag = splat8(bias_g), ao = splat8(bias_o);
#pragma unroll
    for (int kb = 0; kb < 6; ++kb) {
      const v16bf a = (kb < 2) ? load_a_frag(xbuf, kF, kb * 32, lane)
                               : load_a_frag(hcur, kH, (kb - 2) * 32, lane);
      ai = wmma_bf16(a, wgi[kb], ai);
      af = wmma_bf16(a, wgf[kb], af);
      ag = wmma_bf16(a, wgg[kb], ag);
      ao = wmma_bf16(a, wgo[kb], ao);
    }
#pragma unroll
    for (int r = 0; r < 8; ++r) {
      const float iv = sigf(ai[r]);
      const float fv = sigf(af[r]);
      const float gv = FAST_TANH(ag[r]);
      const float ov = sigf(ao[r]);
      const float cn = fv * creg[r] + iv * gv;
      creg[r] = cn;
      const float hn = ov * FAST_TANH(cn);
      const int m = r + hi * 8;                  // D-layout row
      hnxt[m * kH + hcol] = (__bf16)hn;
    }
    __syncthreads();   // hnxt complete; xbuf(t) dead -> safe to rebuild

    // phase B: build x_in for t+1 in-place while computing the dense head.
    //   waves 0-3 : out(t) = h_new x W_den^T -> global out; feedback rows of
    //               xbuf get bf16(out(t)) directly (no staging buffer).
    //   waves 4-7 : non-feedback rows of xbuf get x[:, :, t+1] from global.
    // The two write sets are disjoint by the per-row feedback predicate.
    const int tn = t + 1;
    if (w < 4) {
      v8f da = splat8(dbias);
#pragma unroll
      for (int kb = 0; kb < 4; ++kb) {
        const v16bf a = load_a_frag(hnxt, kH, kb * 32, lane);
        da = wmma_bf16(a, wd[kb], da);
      }
      const int f = w * 16 + n;
#pragma unroll
      for (int r = 0; r < 8; ++r) {
        const int m = r + hi * 8;
        const float v = da[r];
        outg[((size_t)(b0 + m) * kF + f) * kT + t] = v;
        if ((tn >= avail_s[m]) && (tn < pred_s[m]))
          xbuf[m * kF + f] = (__bf16)v;          // autoregressive feedback
      }
    } else if (tn < max_idx) {
      for (int idx = tid - 128; idx < kRows * kF; idx += kNT / 2) {
        const int m = idx >> 6, f = idx & (kF - 1);
        if (!((tn >= avail_s[m]) && (tn < pred_s[m]))) {
          const float* p = xg + ((size_t)(b0 + m) * kF + f) * kT + tn;
          xbuf[idx] = (__bf16)(*p);
          __builtin_prefetch(p + 32, 0, 0);  // next cacheline of this (b,f) row
        }
      }
    }
    __syncthreads();   // xbuf(t+1) complete; hnxt stable for next gates
  }

  // tail: reference zeroes outputs for t >= max_idx
  for (int t = max_idx; t < kT; ++t) {
    for (int idx = tid; idx < kRows * kF; idx += kNT) {
      const int m = idx >> 6, f = idx & (kF - 1);
      outg[((size_t)(b0 + m) * kF + f) * kT + t] = 0.0f;
    }
  }
}

// ---------------- host launch ----------------
extern "C" void kernel_launch(void* const* d_in, const int* in_sizes, int n_in,
                              void* d_out, int out_size, void* d_ws, size_t ws_size,
                              hipStream_t stream) {
  const float* x      = (const float*)d_in[0];
  const int*   avail  = (const int*)  d_in[1];
  const int*   pred   = (const int*)  d_in[2];
  const float* Wih    = (const float*)d_in[3];
  const float* Whh    = (const float*)d_in[4];
  const float* bih    = (const float*)d_in[5];
  const float* bhh    = (const float*)d_in[6];
  const float* Wden   = (const float*)d_in[7];
  const float* bden   = (const float*)d_in[8];
  const float* initch = (const float*)d_in[9];
  float* out = (float*)d_out;

  const size_t smem =
      (size_t)kFrag * 512 * sizeof(__bf16) +        // weight fragments
      2 * (size_t)kRows * kH * sizeof(__bf16) +     // h ping-pong
      (size_t)kRows * kF * sizeof(__bf16) +         // x_in
      (2 * kRows + 2) * sizeof(int);                // lens + reduction slot

  // 320KB LDS per WGP on CDNA5; raise the dynamic-LDS cap (idempotent).
  (void)hipFuncSetAttribute((const void*)lstm_autoreg_kernel,
                            hipFuncAttributeMaxDynamicSharedMemorySize, (int)smem);

  lstm_autoreg_kernel<<<dim3(kB / kRows), dim3(kNT), smem, stream>>>(
      x, avail, pred, Wih, Whh, bih, bhh, Wden, bden, initch, out);
}
